// Policy_46557445488821
// MI455X (gfx1250) — matrix operand order repaired and checked by reading, not compiled, not run on hardware
//
#include <hip/hip_runtime.h>
#include <math.h>

#define DNN 128

typedef __attribute__((ext_vector_type(16))) _Float16 v16h;
typedef __attribute__((ext_vector_type(8)))  float    v8f;

union FragAB { v16h h; unsigned int u[8]; _Float16 e[16]; };

__device__ __forceinline__ void atomicMaxF(float* addr, float v) {
    // race-safe monotone float max via int punning (works with mixed signs, init=-inf)
    if (v >= 0.f) atomicMax((int*)addr, __float_as_int(v));
    else          atomicMin((unsigned int*)addr, (unsigned int)__float_as_int(v));
}
__device__ __forceinline__ float sigm(float x) { return 1.f / (1.f + expf(-x)); }

// ---------------------------------------------------------------------------
// init: reset accumulators every call (graph replays must be deterministic)
// ---------------------------------------------------------------------------
__global__ void init_ws(float* maxacc, float* red) {
    int t = threadIdx.x;
    if (t < 256) maxacc[t] = -INFINITY;
    else if (t == 256) red[0] = -INFINITY;
    else if (t == 257) red[1] = 0.f;
}

// ---------------------------------------------------------------------------
// f32 -> f16 weight conversion (done once per call into workspace)
// wb16: [128][8] = W_basic ; wg16: [2][128][128] = {W_anh, W_enh}
// ---------------------------------------------------------------------------
__global__ __launch_bounds__(256) void prep_f16(const float* __restrict__ Wb,
                                                const float* __restrict__ Wanh,
                                                const float* __restrict__ Wenh,
                                                _Float16* __restrict__ wb16,
                                                _Float16* __restrict__ wg16) {
    int i = blockIdx.x * 256 + threadIdx.x;
    if (i < 1024) wb16[i] = (_Float16)Wb[i];
    int k = i - 1024;
    if (k >= 0 && k < 16384) wg16[k] = (_Float16)Wanh[k];
    int k2 = k - 16384;
    if (k2 >= 0 && k2 < 16384) wg16[16384 + k2] = (_Float16)Wenh[k2];
}

// ---------------------------------------------------------------------------
// Pass 1: e = relu(U @ Wb^T + bb) @ Wg^T + bg ; column max over rows (WMMA)
// block = 128 threads = 4 waves; each wave owns a 16-row M-tile (64 rows/blk)
// ---------------------------------------------------------------------------
__global__ __launch_bounds__(128) void pass1_embed_max(
    const float* __restrict__ unitsA, const float* __restrict__ unitsB, int nA, int nB,
    const _Float16* __restrict__ wb16, const _Float16* __restrict__ wg16,
    const float* __restrict__ b_basic, const float* __restrict__ b_gA,
    const float* __restrict__ b_gB, float* __restrict__ maxacc) {
    __shared__ __align__(16) _Float16     WgL[128][136];   // padded rows: bank spread
    __shared__ __align__(16) unsigned int WbU[128][4];     // packed f16 pairs k=0..7
    __shared__ float bbL[128];
    __shared__ float bgL[128];
    __shared__ __align__(16) float    uL[64][8];
    __shared__ __align__(16) _Float16 hL[4][16][136];      // wave-private h tiles
    __shared__ float wmax[4][128];

    const int tid = threadIdx.x;
    const int grp = blockIdx.y;
    const float*    units = grp ? unitsB : unitsA;
    const int       N     = grp ? nB : nA;
    const _Float16* wg    = wg16 + (size_t)grp * DNN * DNN;
    const float*    bg    = grp ? b_gB : b_gA;

    // stage weights / biases
    for (int i = tid; i < 2048; i += 128) {                // 128 rows x 128 halfs
        int j = i >> 4, kc = (i & 15) << 3;
        *(uint4*)&WgL[j][kc] = *(const uint4*)(wg + j * DNN + kc);
    }
    *(uint4*)&WbU[tid][0] = *(const uint4*)(wb16 + tid * 8);
    bbL[tid] = b_basic[tid];
    bgL[tid] = bg[tid];
    for (int i = tid; i < 4 * 128; i += 128) wmax[i >> 7][i & 127] = -INFINITY;

    // stage unit tile (zero-pad rows >= N so WMMA inputs are defined)
    const int base = blockIdx.x * 64;
    {
        int r = tid >> 1, off = (tid & 1) * 4;
        int gr = base + r;
        float4 v = make_float4(0.f, 0.f, 0.f, 0.f);
        if (gr < N) v = *(const float4*)(units + (size_t)gr * 8 + off);
        *(float4*)&uL[r][off] = v;
    }
    __syncthreads();

    const int w = tid >> 5, lane = tid & 31;
    const int half = lane >> 4, ln = lane & 15;

    // ---- layer 1: h = relu(U @ Wb^T + bb); K padded 8 -> 32 ----
    // A 16x32 f16 layout: lanes0-15 rows; vgpr v holds K=(v>>2)*16+half*8+(v&3)*2
    FragAB a1;
#pragma unroll
    for (int v = 0; v < 8; ++v) a1.u[v] = 0u;
    if (half == 0) {
        const float* up = &uL[w * 16 + ln][0];
#pragma unroll
        for (int v = 0; v < 4; ++v) {                       // K=2v,2v+1 (<8)
            a1.e[2 * v]     = (_Float16)up[2 * v];
            a1.e[2 * v + 1] = (_Float16)up[2 * v + 1];
        }
    }
#pragma unroll
    for (int t = 0; t < 8; ++t) {                           // 8 N-tiles of 16
        FragAB b1;                                          // B[k][n]=Wb[n_g][k]
        b1.u[4] = 0u; b1.u[5] = 0u; b1.u[6] = 0u; b1.u[7] = 0u;
        if (half == 0) {                                    // K=2v<8 only
            *(uint4*)&b1.u[0] = *(const uint4*)&WbU[t * 16 + ln][0];
        } else {
            b1.u[0] = 0u; b1.u[1] = 0u; b1.u[2] = 0u; b1.u[3] = 0u;
        }
        v8f d = {0.f, 0.f, 0.f, 0.f, 0.f, 0.f, 0.f, 0.f};
        d = __builtin_amdgcn_wmma_f32_16x16x32_f16(false, a1.h, false, b1.h,
                                                   (short)0, d, false, false);
        const int   col = t * 16 + ln;
        const float bb  = bbL[col];
#pragma unroll
        for (int v = 0; v < 8; ++v) {                       // D: m=half*8+v, n=col
            float hv = d[v] + bb;
            hL[w][half * 8 + v][col] = (_Float16)(hv > 0.f ? hv : 0.f);
        }
    }
    // hL[w] is wave-private: same-wave DS ordering, no block barrier needed

    // ---- layer 2: e = h @ Wg^T + bg ; per-column max ----
#pragma unroll 1
    for (int t = 0; t < 8; ++t) {
        v8f acc = {0.f, 0.f, 0.f, 0.f, 0.f, 0.f, 0.f, 0.f};
        const _Float16* hrow = &hL[w][ln][0];
        const _Float16* wrow = &WgL[t * 16 + ln][0];
#pragma unroll
        for (int c = 0; c < 4; ++c) {                       // K = 128 in 4 chunks
            FragAB a2, b2;
            // A frag: u[0..3] = halfs [c*32+half*8 .. +7], u[4..7] = +16 (16B aligned)
            *(uint4*)&a2.u[0] = *(const uint4*)(hrow + c * 32 + half * 8);
            *(uint4*)&a2.u[4] = *(const uint4*)(hrow + c * 32 + 16 + half * 8);
            // B frag: u[0..3] = halfs [c*32+half*8 .. +7], u[4..7] = +16 (16B aligned)
            *(uint4*)&b2.u[0] = *(const uint4*)(wrow + c * 32 + half * 8);
            *(uint4*)&b2.u[4] = *(const uint4*)(wrow + c * 32 + 16 + half * 8);
            acc = __builtin_amdgcn_wmma_f32_16x16x32_f16(false, a2.h, false, b2.h,
                                                         (short)0, acc, false, false);
        }
        const int   col = t * 16 + ln;
        const float bgv = bgL[col];
        float mx = -INFINITY;
#pragma unroll
        for (int v = 0; v < 8; ++v) {
            const int gr = base + w * 16 + half * 8 + v;
            const float e = acc[v] + bgv;
            if (gr < N) mx = fmaxf(mx, e);                  // mask padded rows
        }
        mx = fmaxf(mx, __shfl_xor(mx, 16));                 // halves share column
        if (half == 0) wmax[w][col] = fmaxf(wmax[w][col], mx);
    }
    __syncthreads();
    {
        float m = wmax[0][tid];
#pragma unroll
        for (int ww = 1; ww < 4; ++ww) m = fmaxf(m, wmax[ww][tid]);
        atomicMaxF(maxacc + grp * DNN + tid, m);
    }
}

// ---------------------------------------------------------------------------
// Finalize: env head, hero embeddings, pre-layer, LSTM, heads, q, folded r/s
// ---------------------------------------------------------------------------
__global__ __launch_bounds__(128) void finalize_small(
    const float* env, const float* ah, const float* eh,
    const float* h0, const float* c0,
    const float* W_env, const float* b_env,
    const float* W_basic, const float* b_basic,
    const float* W_ah, const float* b_ah,
    const float* W_eh, const float* b_eh,
    const float* W_anh, const float* b_anh,
    const float* W_enh, const float* b_enh,
    const float* W_pre, const float* b_pre,
    const float* W_ih, const float* b_ih,
    const float* W_hh, const float* b_hh,
    const float* W_enum, const float* b_enum,
    const float* W_mx, const float* b_mx,
    const float* W_my, const float* b_my,
    const float* W_att, const float* b_att,
    const float* maxacc, float* red, float* heroE, float* rvec, float* svec,
    float* logits, float* out, int h1Off, int c1Off) {
    __shared__ float s_h[128], s_cat[640], s_x[128], s_h1[128], s_q[128];
    const int j = threadIdx.x;

    { // env embedding
        float e = W_env[j * 2] * env[0] + W_env[j * 2 + 1] * env[1] + b_env[j];
        s_cat[j] = e > 0.f ? e : 0.f;
    }
    // hero groups (tiny N=5): full-precision VALU
    float ahm = -INFINITY, ehm = -INFINITY;
    for (int i = 0; i < 5; ++i) {
        float hv = b_basic[j];
#pragma unroll
        for (int k = 0; k < 8; ++k) hv += W_basic[j * 8 + k] * ah[i * 8 + k];
        s_h[j] = hv > 0.f ? hv : 0.f;
        __syncthreads();
        float e = b_ah[j];
        for (int k = 0; k < 128; ++k) e += W_ah[j * 128 + k] * s_h[k];
        heroE[i * 128 + j] = e;
        ahm = fmaxf(ahm, e);
        __syncthreads();
    }
    for (int i = 0; i < 5; ++i) {
        float hv = b_basic[j];
#pragma unroll
        for (int k = 0; k < 8; ++k) hv += W_basic[j * 8 + k] * eh[i * 8 + k];
        s_h[j] = hv > 0.f ? hv : 0.f;
        __syncthreads();
        float e = b_eh[j];
        for (int k = 0; k < 128; ++k) e += W_eh[j * 128 + k] * s_h[k];
        heroE[(5 + i) * 128 + j] = e;
        ehm = fmaxf(ehm, e);
        __syncthreads();
    }
    s_cat[128 + j] = ahm;
    s_cat[256 + j] = ehm;
    s_cat[384 + j] = maxacc[j];
    s_cat[512 + j] = maxacc[128 + j];
    s_h[j] = h0[j];
    __syncthreads();
    { // pre-layer
        float x = b_pre[j];
        for (int k = 0; k < 640; ++k) x += W_pre[j * 640 + k] * s_cat[k];
        s_x[j] = x > 0.f ? x : 0.f;
    }
    __syncthreads();
    { // LSTM single step, gate order (i,f,g,o)
        float gi = b_ih[j] + b_hh[j];
        float gf = b_ih[128 + j] + b_hh[128 + j];
        float gg = b_ih[256 + j] + b_hh[256 + j];
        float go = b_ih[384 + j] + b_hh[384 + j];
        for (int k = 0; k < 128; ++k) {
            const float xv = s_x[k], hv = s_h[k];
            gi += W_ih[j * 128 + k] * xv + W_hh[j * 128 + k] * hv;
            gf += W_ih[(128 + j) * 128 + k] * xv + W_hh[(128 + j) * 128 + k] * hv;
            gg += W_ih[(256 + j) * 128 + k] * xv + W_hh[(256 + j) * 128 + k] * hv;
            go += W_ih[(384 + j) * 128 + k] * xv + W_hh[(384 + j) * 128 + k] * hv;
        }
        float c1 = sigm(gf) * c0[j] + sigm(gi) * tanhf(gg);
        float h1 = sigm(go) * tanhf(c1);
        s_h1[j] = h1;
        out[h1Off + j] = h1;
        out[c1Off + j] = c1;
    }
    __syncthreads();
    { // q = W_att h1 + b_att
        float q = b_att[j];
        for (int k = 0; k < 128; ++k) q += W_att[j * 128 + k] * s_h1[k];
        s_q[j] = q;
    }
    __syncthreads();
    // small heads + softmax (serial, tiny)
    if (j == 0) {
        float l[3], m = -INFINITY;
        for (int t = 0; t < 3; ++t) {
            float a = b_enum[t];
            for (int k = 0; k < 128; ++k) a += W_enum[t * 128 + k] * s_h1[k];
            l[t] = a; m = fmaxf(m, a);
        }
        float s = 0.f;
        for (int t = 0; t < 3; ++t) { l[t] = expf(l[t] - m); s += l[t]; }
        for (int t = 0; t < 3; ++t) out[t] = l[t] / s;
    } else if (j == 1) {
        float l[9], m = -INFINITY;
        for (int t = 0; t < 9; ++t) {
            float a = b_mx[t];
            for (int k = 0; k < 128; ++k) a += W_mx[t * 128 + k] * s_h1[k];
            l[t] = a; m = fmaxf(m, a);
        }
        float s = 0.f;
        for (int t = 0; t < 9; ++t) { l[t] = expf(l[t] - m); s += l[t]; }
        for (int t = 0; t < 9; ++t) out[3 + t] = l[t] / s;
    } else if (j == 2) {
        float l[9], m = -INFINITY;
        for (int t = 0; t < 9; ++t) {
            float a = b_my[t];
            for (int k = 0; k < 128; ++k) a += W_my[t * 128 + k] * s_h1[k];
            l[t] = a; m = fmaxf(m, a);
        }
        float s = 0.f;
        for (int t = 0; t < 9; ++t) { l[t] = expf(l[t] - m); s += l[t]; }
        for (int t = 0; t < 9; ++t) out[12 + t] = l[t] / s;
    }
    { // folded attention: r = Wg^T q, s = q . b_g  (anh, enh)
        float r0 = 0.f, r1 = 0.f;
        for (int m = 0; m < 128; ++m) {
            const float qm = s_q[m];
            r0 += qm * W_anh[m * 128 + j];
            r1 += qm * W_enh[m * 128 + j];
        }
        rvec[j] = r0; rvec[128 + j] = r1;
    }
    if (j == 3) { float s = 0.f; for (int m = 0; m < 128; ++m) s += s_q[m] * b_anh[m]; svec[0] = s; }
    if (j == 4) { float s = 0.f; for (int m = 0; m < 128; ++m) s += s_q[m] * b_enh[m]; svec[1] = s; }
    if (j < 10) { // hero logits (embeddings stored above, visible after barriers)
        float l = 0.f;
        for (int k = 0; k < 128; ++k) l += s_q[k] * heroE[j * 128 + k];
        logits[j] = l;
        atomicMaxF(&red[0], l);
    }
}

// ---------------------------------------------------------------------------
// Pass 2: logit = r . relu(Wb u + bb) + s  (thread per unit) + running max
// ---------------------------------------------------------------------------
__global__ __launch_bounds__(256) void pass2_logits(
    const float* __restrict__ unitsA, const float* __restrict__ unitsB, int nA, int nB,
    const float* __restrict__ W_basic, const float* __restrict__ b_basic,
    const float* __restrict__ rvec, const float* __restrict__ svec,
    float* __restrict__ logits, float* __restrict__ red) {
    __shared__ float WbL[128][8];
    __shared__ float bbL[128];
    __shared__ float rL[128];
    __shared__ float smax[256];
    const int tid = threadIdx.x;
    const int grp = blockIdx.y;
    const float* units = grp ? unitsB : unitsA;
    const int N = grp ? nB : nA;
    for (int i = tid; i < 1024; i += 256) WbL[i >> 3][i & 7] = W_basic[i];
    if (tid < 128) { bbL[tid] = b_basic[tid]; rL[tid] = rvec[grp * 128 + tid]; }
    __syncthreads();
    const int idx = blockIdx.x * 256 + tid;
    float logit = -INFINITY;
    if (idx < N) {
        float u[8];
        const float* up = units + (size_t)idx * 8;
        // pull next block's rows toward the cache hierarchy (global_prefetch_b8)
        if (idx + 256 < N) __builtin_prefetch(up + 2048, 0, 1);
#pragma unroll
        for (int k = 0; k < 8; ++k) u[k] = up[k];
        float acc = svec[grp];
        for (int jj = 0; jj < 128; ++jj) {
            float hv = bbL[jj];
#pragma unroll
            for (int k = 0; k < 8; ++k) hv = fmaf(WbL[jj][k], u[k], hv);
            acc = fmaf(rL[jj], fmaxf(hv, 0.f), acc);
        }
        logit = acc;
        logits[10 + (grp ? nA : 0) + idx] = acc;
    }
    smax[tid] = logit;
    __syncthreads();
    for (int s = 128; s > 0; s >>= 1) {
        if (tid < s) smax[tid] = fmaxf(smax[tid], smax[tid + s]);
        __syncthreads();
    }
    if (tid == 0) atomicMaxF(&red[0], smax[0]);
}

// ---------------------------------------------------------------------------
// Softmax over 400010 logits (deterministic two-stage sum)
// ---------------------------------------------------------------------------
__global__ __launch_bounds__(256) void exp_partial(const float* __restrict__ logits,
                                                   const float* __restrict__ red,
                                                   float* __restrict__ psum, int n) {
    __shared__ float s[256];
    const int idx = blockIdx.x * 256 + threadIdx.x;
    const float m = red[0];
    s[threadIdx.x] = (idx < n) ? expf(logits[idx] - m) : 0.f;
    __syncthreads();
    for (int st = 128; st > 0; st >>= 1) {
        if (threadIdx.x < st) s[threadIdx.x] += s[threadIdx.x + st];
        __syncthreads();
    }
    if (threadIdx.x == 0) psum[blockIdx.x] = s[0];
}

__global__ __launch_bounds__(256) void sum_reduce(const float* __restrict__ psum,
                                                  int nblocks, float* __restrict__ red) {
    __shared__ float s[256];
    float acc = 0.f;
    for (int i = threadIdx.x; i < nblocks; i += 256) acc += psum[i];
    s[threadIdx.x] = acc;
    __syncthreads();
    for (int st = 128; st > 0; st >>= 1) {
        if (threadIdx.x < st) s[threadIdx.x] += s[threadIdx.x + st];
        __syncthreads();
    }
    if (threadIdx.x == 0) red[1] = s[0];
}

__global__ __launch_bounds__(256) void normalize_tgt(const float* __restrict__ logits,
                                                     const float* __restrict__ red,
                                                     float* __restrict__ outTgt, int n) {
    const int idx = blockIdx.x * 256 + threadIdx.x;
    if (idx < n) outTgt[idx] = expf(logits[idx] - red[0]) / red[1];
}

// ---------------------------------------------------------------------------
extern "C" void kernel_launch(void* const* d_in, const int* in_sizes, int n_in,
                              void* d_out, int out_size, void* d_ws, size_t ws_size,
                              hipStream_t stream) {
    (void)n_in; (void)out_size; (void)ws_size;
    const float* env     = (const float*)d_in[0];
    const float* ah      = (const float*)d_in[1];
    const float* eh      = (const float*)d_in[2];
    const float* anh     = (const float*)d_in[3];
    const float* enh     = (const float*)d_in[4];
    const float* h0      = (const float*)d_in[5];
    const float* c0      = (const float*)d_in[6];
    const float* W_env   = (const float*)d_in[7];  const float* b_env  = (const float*)d_in[8];
    const float* W_basic = (const float*)d_in[9];  const float* b_basic= (const float*)d_in[10];
    const float* W_ah    = (const float*)d_in[11]; const float* b_ah   = (const float*)d_in[12];
    const float* W_eh    = (const float*)d_in[13]; const float* b_eh   = (const float*)d_in[14];
    const float* W_anh   = (const float*)d_in[15]; const float* b_anh  = (const float*)d_in[16];
    const float* W_enh   = (const float*)d_in[17]; const float* b_enh  = (const float*)d_in[18];
    const float* W_pre   = (const float*)d_in[19]; const float* b_pre  = (const float*)d_in[20];
    const float* W_ih    = (const float*)d_in[21]; const float* b_ih   = (const float*)d_in[22];
    const float* W_hh    = (const float*)d_in[23]; const float* b_hh   = (const float*)d_in[24];
    const float* W_enum  = (const float*)d_in[25]; const float* b_enum = (const float*)d_in[26];
    const float* W_mx    = (const float*)d_in[27]; const float* b_mx   = (const float*)d_in[28];
    const float* W_my    = (const float*)d_in[29]; const float* b_my   = (const float*)d_in[30];
    const float* W_att   = (const float*)d_in[31]; const float* b_att  = (const float*)d_in[32];
    float* out = (float*)d_out;

    const int nA   = in_sizes[3] / 8;     // 200000
    const int nB   = in_sizes[4] / 8;     // 200000
    const int nTot = 10 + nA + nB;        // 400010

    // workspace layout (floats); f16 region aligned to 16 bytes for uint4 loads
    float* ws     = (float*)d_ws;
    float* maxacc = ws;                    // [2][128]
    float* red    = ws + 256;              // [2]: logit max, exp-sum
    float* heroE  = ws + 260;              // [10][128]
    float* rvec   = ws + 1540;             // [2][128]
    float* svec   = ws + 1796;             // [2]
    float* logits = ws + 1800;             // [nTot]
    size_t f16off = ((size_t)(1800 + nTot) + 3) & ~(size_t)3;
    _Float16* wb16 = (_Float16*)(ws + f16off);   // [128][8]
    _Float16* wg16 = wb16 + 1024;                // [2][128][128]
    float*    psum = (float*)(wg16 + 2 * DNN * DNN);

    init_ws<<<1, 320, 0, stream>>>(maxacc, red);
    prep_f16<<<(1024 + 2 * DNN * DNN + 255) / 256, 256, 0, stream>>>(
        W_basic, W_anh, W_enh, wb16, wg16);

    const int nMax = nA > nB ? nA : nB;
    dim3 g1((nMax + 63) / 64, 2);
    pass1_embed_max<<<g1, 128, 0, stream>>>(anh, enh, nA, nB, wb16, wg16,
                                            b_basic, b_anh, b_enh, maxacc);

    finalize_small<<<1, 128, 0, stream>>>(
        env, ah, eh, h0, c0, W_env, b_env, W_basic, b_basic, W_ah, b_ah, W_eh, b_eh,
        W_anh, b_anh, W_enh, b_enh, W_pre, b_pre, W_ih, b_ih, W_hh, b_hh,
        W_enum, b_enum, W_mx, b_mx, W_my, b_my, W_att, b_att,
        maxacc, red, heroE, rvec, svec, logits, out, 21 + nTot, 21 + nTot + DNN);

    dim3 g2((nMax + 255) / 256, 2);
    pass2_logits<<<g2, 256, 0, stream>>>(anh, enh, nA, nB, W_basic, b_basic,
                                         rvec, svec, logits, red);

    const int nbE = (nTot + 255) / 256;
    exp_partial<<<nbE, 256, 0, stream>>>(logits, red, psum, nTot);
    sum_reduce<<<1, 256, 0, stream>>>(psum, nbE, red);
    normalize_tgt<<<nbE, 256, 0, stream>>>(logits, red, out + 21, nTot);
}
